// MultiheadAttention_86268713108065
// MI455X (gfx1250) — matrix operand order repaired and checked by reading, not compiled, not run on hardware
//
#include <hip/hip_runtime.h>
#include <hip/hip_bf16.h>
#include <cstdint>

// ---------------------------------------------------------------------------
// Problem constants (ViT-base attention with iRPE)
// ---------------------------------------------------------------------------
constexpr int Bb  = 64;          // batch
constexpr int Nn  = 197;         // tokens
constexpr int Dd  = 768;         // model dim
constexpr int Hh  = 12;          // heads
constexpr int HD  = 64;          // head dim
constexpr int BH  = Bb * Hh;     // 768
constexpr int NP  = 224;         // tokens padded to 7*32 (WMMA tiling)
constexpr int BN  = Bb * Nn;     // 12608 = 394 * 32 (exactly tileable)
constexpr float SCALE = 0.125f;  // 64^-0.5

typedef __attribute__((ext_vector_type(16))) __bf16 v16bf;
typedef __attribute__((ext_vector_type(8)))  __bf16 bf16x8;
typedef __attribute__((ext_vector_type(8)))  float  v8f;

// ---------------------------------------------------------------------------
// 32x32 register-blocked WMMA tile (2x2 grid of 16x16 WMMAs per wave):
//   C[row0:+32, col0:+32] = sum_k A(r,k)*B(k,c)
// Each A/B fragment is reused by two WMMAs -> 128B of loads per 4 WMMAs.
// Fragment layouts used here (A/B share contiguous k order, not the ISA table order):
//   A 16x32 bf16 : lane l -> row M=l%16 ; elem e -> K = 16*(l/16)+e
//   B 32x16 bf16 : lane l -> col N=l%16 ; elem e -> K = 16*(l/16)+e
//   C/D 16x16 f32: vgpr r -> M = r + 8*(l/16), N = l%16
// ptrA(r,k): pointer to A(r,k); A(r, k..k+7) must be contiguous.
// ptrB(k,c): pointer to B(k,c); B(k..k+15, c) must be contiguous (B^T layout).
// ---------------------------------------------------------------------------
#define SHUF16(x, y) __builtin_shufflevector(x, y, 0, 1, 2, 3, 4, 5, 6, 7, \
                                             8, 9, 10, 11, 12, 13, 14, 15)

template <typename AP, typename BP, typename CS>
__device__ __forceinline__ void wmma_tile32(int row0, int col0, int K,
                                            AP ptrA, BP ptrB, CS store) {
  const int lane = threadIdx.x & 31;
  const int half = lane >> 4;   // 0 or 1
  const int lm   = lane & 15;
  const int rA0  = row0 + lm,      rA1 = row0 + 16 + lm;
  const int cB0  = col0 + lm,      cB1 = col0 + 16 + lm;
  v8f acc00 = {}, acc01 = {}, acc10 = {}, acc11 = {};
  for (int k0 = 0; k0 < K; k0 += 32) {
    const int kA = k0 + (half << 4);
    const int kB = k0 + (half << 4);
    const bf16x8 a00 = *(const bf16x8*)ptrA(rA0, kA);
    const bf16x8 a01 = *(const bf16x8*)ptrA(rA0, kA + 8);
    const bf16x8 a10 = *(const bf16x8*)ptrA(rA1, kA);
    const bf16x8 a11 = *(const bf16x8*)ptrA(rA1, kA + 8);
    const v16bf  aF0 = SHUF16(a00, a01);
    const v16bf  aF1 = SHUF16(a10, a11);
    const v16bf  b0  = *(const v16bf*)ptrB(kB, cB0);
    const v16bf  b1  = *(const v16bf*)ptrB(kB, cB1);
    if (k0 + 32 < K) {   // prefetch next k-step (global_prefetch_b8)
      __builtin_prefetch(ptrA(rA0, kA + 32), 0, 1);
      __builtin_prefetch(ptrA(rA1, kA + 32), 0, 1);
      __builtin_prefetch(ptrB(kB + 32, cB0), 0, 1);
      __builtin_prefetch(ptrB(kB + 32, cB1), 0, 1);
    }
    acc00 = __builtin_amdgcn_wmma_f32_16x16x32_bf16(false, aF0, false, b0,
                                                    (short)0, acc00, false, false);
    acc01 = __builtin_amdgcn_wmma_f32_16x16x32_bf16(false, aF0, false, b1,
                                                    (short)0, acc01, false, false);
    acc10 = __builtin_amdgcn_wmma_f32_16x16x32_bf16(false, aF1, false, b0,
                                                    (short)0, acc10, false, false);
    acc11 = __builtin_amdgcn_wmma_f32_16x16x32_bf16(false, aF1, false, b1,
                                                    (short)0, acc11, false, false);
  }
#pragma unroll
  for (int r = 0; r < 8; ++r) {
    const int m0 = row0 + (half << 3) + r;
    store(m0,      col0 + lm,      acc00[r]);
    store(m0,      col0 + 16 + lm, acc01[r]);
    store(m0 + 16, col0 + lm,      acc10[r]);
    store(m0 + 16, col0 + 16 + lm, acc11[r]);
  }
}

// ---------------------------------------------------------------------------
// Elementwise helpers
// ---------------------------------------------------------------------------
__global__ void k_f32_to_bf16(const float* __restrict__ s,
                              __bf16* __restrict__ d, int n) {
  int stride = gridDim.x * blockDim.x;
  for (int i = blockIdx.x * blockDim.x + threadIdx.x; i < n; i += stride)
    d[i] = (__bf16)s[i];
}

// transpose-convert a Dd x Dd weight: w[k][c] (f32) -> wT[c][k] (bf16)
__global__ void k_w_to_bf16T(const float* __restrict__ w, __bf16* __restrict__ wT) {
  int stride = gridDim.x * blockDim.x;
  for (int i = blockIdx.x * blockDim.x + threadIdx.x; i < Dd * Dd; i += stride) {
    const int k = i / Dd, c = i - k * Dd;
    wT[(size_t)c * Dd + k] = (__bf16)w[i];
  }
}

__global__ void k_zero_bf16(__bf16* __restrict__ p, int n) {
  int stride = gridDim.x * blockDim.x;
  for (int i = blockIdx.x * blockDim.x + threadIdx.x; i < n; i += stride)
    p[i] = (__bf16)0.0f;
}

// ---------------------------------------------------------------------------
// Build iRPE tables on device (IDX_V/IDX_H are not kernel inputs):
//   rpk [i][j][d] = bf16(rel_k_v[iv][d] + rel_k_h[ih][d])   (NP x 64, d-contig)
//   rpvT[i][d][j] = bf16(rel_v_v[iv][d] + rel_v_h[ih][d])   (64 x NP, j-contig)
// j in [197,224) zero-filled so padded contractions are exact.
// grid.x = 197 (i), blockDim.x = 224 (j)
// ---------------------------------------------------------------------------
__global__ void k_rpe(const float* __restrict__ rkv, const float* __restrict__ rkh,
                      const float* __restrict__ rvv, const float* __restrict__ rvh,
                      __bf16* __restrict__ rpk, __bf16* __restrict__ rpvT) {
  const int i = blockIdx.x;
  const int j = threadIdx.x;
  __bf16* pk = rpk  + ((size_t)i * NP + j) * 64;     // [j][d]
  __bf16* pv = rpvT + (size_t)i * 64 * NP;           // [d][j]
  if (j >= Nn) {
    for (int d = 0; d < 64; ++d) { pk[d] = (__bf16)0.0f; pv[d * NP + j] = (__bf16)0.0f; }
    return;
  }
  int iv = 0, ih = 0;
  if (i > 0 && j > 0) {
    const int ri = i - 1, rj = j - 1;
    int dv = rj / 14 - ri / 14;
    int dh = rj % 14 - ri % 14;
    dv = dv < -14 ? -14 : (dv > 14 ? 14 : dv);
    dh = dh < -14 ? -14 : (dh > 14 ? 14 : dh);
    iv = dv + 15; ih = dh + 15;
  }
  for (int d = 0; d < 64; ++d) {
    pk[d]          = (__bf16)(rkv[iv * 64 + d] + rkh[ih * 64 + d]);
    pv[d * NP + j] = (__bf16)(rvv[iv * 64 + d] + rvh[ih * 64 + d]);
  }
}

// ---------------------------------------------------------------------------
// QKV projection: Y = x @ W + bias, scaled, scattered to per-head layout.
//   vtrans=0: out[bh][n][d]  (Q, K)      vtrans=1: out[bh][d][n]  (V^T)
// grid(394, 6), block 128 (4 waves -> 4 adjacent 32-col tiles). K = 768.
// ---------------------------------------------------------------------------
__global__ void k_gemm_qkv(const __bf16* __restrict__ xb, const __bf16* __restrict__ wT,
                           const float* __restrict__ bias, __bf16* __restrict__ out,
                           float scl, int vtrans) {
  const int row0 = blockIdx.x * 32;
  const int col0 = blockIdx.y * 128 + (threadIdx.x >> 5) * 32;
  wmma_tile32(row0, col0, Dd,
    [&](int r, int k) { return xb + (size_t)r * Dd + k; },
    [&](int k, int c) { return wT + (size_t)c * Dd + k; },
    [&](int r, int c, float v) {
      const int b = r / Nn, n = r - b * Nn;
      const int h = c >> 6, d = c & 63;
      const size_t idx = vtrans
          ? (((size_t)(b * Hh + h)) * 64 + d) * NP + n
          : (((size_t)(b * Hh + h)) * NP + n) * 64 + d;
      out[idx] = (__bf16)((v + bias[c]) * scl);
    });
}

// ---------------------------------------------------------------------------
// S1 = (scaled Q) @ K^T per (b,h).  K rows are already the B^T layout.
// grid(768, 7, 7), block 32 (one wave, 32x32 tile). K = 64.
// ---------------------------------------------------------------------------
__global__ void k_score1(const __bf16* __restrict__ Qb, const __bf16* __restrict__ Kb,
                         __bf16* __restrict__ Sb) {
  const size_t bh = blockIdx.x;
  const __bf16* Q  = Qb + bh * NP * 64;
  const __bf16* Kp = Kb + bh * NP * 64;
  __bf16* S = Sb + bh * (size_t)NP * NP;
  wmma_tile32(blockIdx.y * 32, blockIdx.z * 32, 64,
    [&](int r, int k) { return Q + (size_t)r * 64 + k; },
    [&](int k, int c) { return Kp + (size_t)c * 64 + k; },
    [&](int r, int c, float v) { S[(size_t)r * NP + c] = (__bf16)v; });
}

// ---------------------------------------------------------------------------
// S += per-query-index rpk GEMM: S[:, i, :] += Q[:, i, :] @ rpk[i]^T.
// rpk stored [i][j][d] so columns are k-contiguous.
// grid(197, 24, 7), block 32 (one wave, 32x32 tile). K = 64.
// ---------------------------------------------------------------------------
__global__ void k_score2(const __bf16* __restrict__ Qb, const __bf16* __restrict__ rpk,
                         __bf16* __restrict__ Sb) {
  const int i = blockIdx.x;
  const __bf16* Bp = rpk + (size_t)i * NP * 64;
  wmma_tile32(blockIdx.y * 32, blockIdx.z * 32, 64,
    [&](int r, int k) { return Qb + ((size_t)r * NP + i) * 64 + k; },
    [&](int k, int c) { return Bp + (size_t)c * 64 + k; },
    [&](int r, int c, float v) {
      const size_t idx = ((size_t)r * NP + i) * NP + c;
      Sb[idx] = (__bf16)(v + (float)Sb[idx]);
    });
}

// ---------------------------------------------------------------------------
// Row softmax in f32, masked to j<197, zero-filling the padded columns.
// grid(768, 197), block 32 (one wave32 per row, 7 cols/lane).
// ---------------------------------------------------------------------------
__global__ void k_softmax(__bf16* __restrict__ Sb) {
  const size_t bh = blockIdx.x;
  const int i = blockIdx.y;
  __bf16* row = Sb + (bh * NP + i) * (size_t)NP;
  const int lane = threadIdx.x;
  float x[7];
  float mx = -3.4e38f;
#pragma unroll
  for (int t = 0; t < 7; ++t) {
    const int j = lane + 32 * t;
    x[t] = (j < Nn) ? (float)row[j] : -3.4e38f;
    mx = fmaxf(mx, x[t]);
  }
  for (int o = 16; o > 0; o >>= 1) mx = fmaxf(mx, __shfl_xor(mx, o, 32));
  float sum = 0.0f;
#pragma unroll
  for (int t = 0; t < 7; ++t) {
    const int j = lane + 32 * t;
    x[t] = (j < Nn) ? __expf(x[t] - mx) : 0.0f;
    sum += x[t];
  }
  for (int o = 16; o > 0; o >>= 1) sum += __shfl_xor(sum, o, 32);
  const float inv = 1.0f / sum;
#pragma unroll
  for (int t = 0; t < 7; ++t) row[lane + 32 * t] = (__bf16)(x[t] * inv);
}

// ---------------------------------------------------------------------------
// O1 = attn @ V per (b,h); V stored transposed [bh][d][n] (k-contiguous).
// grid(768, 7), block 64 (2 waves -> 2 x 32 d cols). K = 224.
// ---------------------------------------------------------------------------
__global__ void k_av(const __bf16* __restrict__ Ab, const __bf16* __restrict__ VT,
                     float* __restrict__ Of) {
  const size_t bh = blockIdx.x;
  const __bf16* A = Ab + bh * (size_t)NP * NP;
  const __bf16* V = VT + bh * (size_t)NP * 64;
  float* O = Of + bh * (size_t)NP * 64;
  const int col0 = (threadIdx.x >> 5) * 32;
  wmma_tile32(blockIdx.y * 32, col0, NP,
    [&](int r, int k) { return A + (size_t)r * NP + k; },
    [&](int k, int c) { return V + (size_t)c * NP + k; },
    [&](int r, int c, float v) { O[(size_t)r * 64 + c] = v; });
}

// ---------------------------------------------------------------------------
// O += per-query-index rpv GEMM: O[:, i, :] += attn[:, i, :] @ rpv[i].
// rpv stored transposed [i][d][j] (k-contiguous).
// grid(197, 24), block 64 (2 waves -> 2 x 32 d cols). K = 224.
// ---------------------------------------------------------------------------
__global__ void k_av2(const __bf16* __restrict__ Ab, const __bf16* __restrict__ rpvT,
                      float* __restrict__ Of) {
  const int i = blockIdx.x;
  const __bf16* Bp = rpvT + (size_t)i * 64 * NP;
  const int col0 = (threadIdx.x >> 5) * 32;
  wmma_tile32(blockIdx.y * 32, col0, NP,
    [&](int r, int k) { return Ab + ((size_t)r * NP + i) * NP + k; },
    [&](int k, int c) { return Bp + (size_t)c * NP + k; },
    [&](int r, int c, float v) { Of[((size_t)r * NP + i) * 64 + c] += v; });
}

// ---------------------------------------------------------------------------
// Output projection: out = O @ wp + bp, f32 result.
// grid(394, 6), block 128 (4 waves -> 4 x 32 col tiles). K = 768.
// ---------------------------------------------------------------------------
__global__ void k_proj(const __bf16* __restrict__ Ob, const __bf16* __restrict__ wpT,
                       const float* __restrict__ bp, float* __restrict__ out) {
  const int row0 = blockIdx.x * 32;
  const int col0 = blockIdx.y * 128 + (threadIdx.x >> 5) * 32;
  wmma_tile32(row0, col0, Dd,
    [&](int r, int k) {
      const int b = r / Nn, n = r - b * Nn;
      const int h = k >> 6, d = k & 63;
      return Ob + (((size_t)(b * Hh + h)) * NP + n) * 64 + d;
    },
    [&](int k, int c) { return wpT + (size_t)c * Dd + k; },
    [&](int r, int c, float v) { out[(size_t)r * Dd + c] = v + bp[c]; });
}

// ---------------------------------------------------------------------------
// Host-side launcher
// ---------------------------------------------------------------------------
extern "C" void kernel_launch(void* const* d_in, const int* in_sizes, int n_in,
                              void* d_out, int out_size, void* d_ws, size_t ws_size,
                              hipStream_t stream) {
  const float* x   = (const float*)d_in[0];
  const float* wq  = (const float*)d_in[1];
  const float* bq  = (const float*)d_in[2];
  const float* wk  = (const float*)d_in[3];
  const float* bk  = (const float*)d_in[4];
  const float* wv  = (const float*)d_in[5];
  const float* bv  = (const float*)d_in[6];
  const float* wp  = (const float*)d_in[7];
  const float* bp  = (const float*)d_in[8];
  const float* rkv = (const float*)d_in[9];
  const float* rkh = (const float*)d_in[10];
  const float* rvv = (const float*)d_in[11];
  const float* rvh = (const float*)d_in[12];
  float* out = (float*)d_out;

  // Workspace carve-up (256B aligned)
  char* ws = (char*)d_ws;
  auto take = [&](size_t bytes) {
    char* p = ws;
    ws += (bytes + 255) & ~(size_t)255;
    return p;
  };
  const int nX   = BN * Dd;            // 9,682,944
  const int nW   = Dd * Dd;            // 589,824
  const int nQKV = BH * NP * HD;       // 11,010,048
  const int nRPK = Nn * HD * NP;       // 2,824,192
  const int nS   = BH * NP * NP;       // 38,535,168

  __bf16* xbf  = (__bf16*)take((size_t)nX   * 2);
  __bf16* wqT  = (__bf16*)take((size_t)nW   * 2);
  __bf16* wkT  = (__bf16*)take((size_t)nW   * 2);
  __bf16* wvT  = (__bf16*)take((size_t)nW   * 2);
  __bf16* wpT  = (__bf16*)take((size_t)nW   * 2);
  __bf16* Qbf  = (__bf16*)take((size_t)nQKV * 2);   // [bh][n][d]
  __bf16* Kbf  = (__bf16*)take((size_t)nQKV * 2);   // [bh][n][d]
  __bf16* VbfT = (__bf16*)take((size_t)nQKV * 2);   // [bh][d][n]
  __bf16* rpk  = (__bf16*)take((size_t)nRPK * 2);   // [i][j][d]
  __bf16* rpvT = (__bf16*)take((size_t)nRPK * 2);   // [i][d][j]
  __bf16* Sbf  = (__bf16*)take((size_t)nS   * 2);   // [bh][i][j]
  float*  Of   = (float* )take((size_t)nQKV * 4);   // [bh][n][d]
  __bf16* Obf  = (__bf16*)take((size_t)nQKV * 2);   // [bh][n][d]
  (void)ws_size; (void)in_sizes; (void)n_in; (void)out_size;

  // 1. fp32 -> bf16 conversions (weights transposed for k-contiguous B loads)
  k_f32_to_bf16<<<2048, 256, 0, stream>>>(x, xbf, nX);
  k_w_to_bf16T<<<512, 256, 0, stream>>>(wq, wqT);
  k_w_to_bf16T<<<512, 256, 0, stream>>>(wk, wkT);
  k_w_to_bf16T<<<512, 256, 0, stream>>>(wv, wvT);
  k_w_to_bf16T<<<512, 256, 0, stream>>>(wp, wpT);

  // 2. zero Q/K/V so padded token rows/cols contribute exact zeros
  k_zero_bf16<<<2048, 256, 0, stream>>>(Qbf, nQKV);
  k_zero_bf16<<<2048, 256, 0, stream>>>(Kbf, nQKV);
  k_zero_bf16<<<2048, 256, 0, stream>>>(VbfT, nQKV);

  // 3. iRPE tables (indices recomputed on device)
  k_rpe<<<Nn, NP, 0, stream>>>(rkv, rkh, rvv, rvh, rpk, rpvT);

  // 4. QKV projections (SCALE folded into Q; V written transposed)
  dim3 gq(BN / 32, Dd / 128);
  k_gemm_qkv<<<gq, 128, 0, stream>>>(xbf, wqT, bq, Qbf,  SCALE, 0);
  k_gemm_qkv<<<gq, 128, 0, stream>>>(xbf, wkT, bk, Kbf,  1.0f,  0);
  k_gemm_qkv<<<gq, 128, 0, stream>>>(xbf, wvT, bv, VbfT, 1.0f,  1);

  // 5. attention scores: S1 = QK^T, then S += per-i rpk GEMM
  k_score1<<<dim3(BH, NP / 32, NP / 32), 32, 0, stream>>>(Qbf, Kbf, Sbf);
  k_score2<<<dim3(Nn, BH / 32, NP / 32), 32, 0, stream>>>(Qbf, rpk, Sbf);

  // 6. masked softmax (f32 math, zero-filled pad columns)
  k_softmax<<<dim3(BH, Nn), 32, 0, stream>>>(Sbf);

  // 7. O = attn @ V, then O += per-i rpv GEMM
  k_av <<<dim3(BH, NP / 32), 64, 0, stream>>>(Sbf, VbfT, Of);
  k_av2<<<dim3(Nn, BH / 32), 64, 0, stream>>>(Sbf, rpvT, Of);

  // 8. convert O to bf16 and run the output projection
  k_f32_to_bf16<<<2048, 256, 0, stream>>>(Of, Obf, nQKV);
  k_proj<<<dim3(BN / 32, Dd / 128), 128, 0, stream>>>(Obf, wpT, bp, out);
}
